// Feature_Decoder_47029891891861
// MI455X (gfx1250) — matrix operand order repaired and checked by reading, not compiled, not run on hardware
//
#include <hip/hip_runtime.h>
#include <hip/hip_bf16.h>
#include <cstdint>
#include <cstddef>

// ---------------------------------------------------------------------------
// Sizes (fixed by the reference)
// ---------------------------------------------------------------------------
#define D_MODEL 1024
#define N_FEAT  512
#define N_CLUST 256
#define N_SEQ   2048
#define N_BATCH 32
#define LN_EPS  1e-5f
#define NEG_SLOPE 0.2f

typedef __attribute__((ext_vector_type(16))) __bf16 v16bf;
typedef __attribute__((ext_vector_type(8)))  float  v8f;

// f32 -> bf16 round-to-nearest-even
__device__ __forceinline__ unsigned short f2bf(float x) {
    unsigned int b = __float_as_uint(x);
    b += 0x7FFFu + ((b >> 16) & 1u);
    return (unsigned short)(b >> 16);
}

union FragB {
    v16bf v;
    unsigned short u[16];
    float4 f4[2];
};

__device__ __forceinline__ void cvt8(FragB& f, int base, float4 a, float4 b) {
    f.u[base + 0] = f2bf(a.x); f.u[base + 1] = f2bf(a.y);
    f.u[base + 2] = f2bf(a.z); f.u[base + 3] = f2bf(a.w);
    f.u[base + 4] = f2bf(b.x); f.u[base + 5] = f2bf(b.y);
    f.u[base + 6] = f2bf(b.z); f.u[base + 7] = f2bf(b.w);
}

// f32-source A fragment with on-the-fly convert (used by the small gate GEMM)
__device__ __forceinline__ void load_afrag_f32(FragB& a, const float* __restrict__ arow,
                                               int k, int half) {
    const int klo = k + half * 8;
    float4 p0 = *(const float4*)(arow + klo);
    float4 p1 = *(const float4*)(arow + klo + 4);
    float4 p2 = *(const float4*)(arow + klo + 16);
    float4 p3 = *(const float4*)(arow + klo + 20);
    cvt8(a, 0, p0, p1);
    cvt8(a, 8, p2, p3);
}

// ---------------------------------------------------------------------------
// Kernel 1: elementwise f32 -> bf16 (weights only)
// ---------------------------------------------------------------------------
__global__ void convert_f32_bf16(const float* __restrict__ src,
                                 unsigned short* __restrict__ dst, int n) {
    int i = blockIdx.x * blockDim.x + threadIdx.x;
    if (i < n) dst[i] = f2bf(src[i]);
}

// ---------------------------------------------------------------------------
// Kernel 2: gate GEMM  c[n,d] = sum_k static[n,k] * W_city[d,k] + b_city[d]
// ---------------------------------------------------------------------------
__global__ void __launch_bounds__(256)
city_gemm(const float* __restrict__ statc, const unsigned short* __restrict__ wbf,
          const float* __restrict__ bias, float* __restrict__ cout) {
    const int tid  = threadIdx.x;
    const int wave = tid >> 5, lane = tid & 31;
    const int half = lane >> 4, l16 = lane & 15;
    const int colBase = wave * 128;
    const int rowBase = blockIdx.x * 16;

    v8f acc[8];
#pragma unroll
    for (int t = 0; t < 8; ++t) acc[t] = {};

    const float* arow = statc + (size_t)(rowBase + l16) * N_CLUST;
    for (int k = 0; k < N_CLUST; k += 32) {
        FragB a;
        load_afrag_f32(a, arow, k, half);
        const int kb = k + half * 8;
        FragB bb[8];
#pragma unroll
        for (int t = 0; t < 8; ++t) {
            const float4* bp = (const float4*)(wbf + (size_t)(colBase + t * 16 + l16) * N_CLUST + kb);
            bb[t].f4[0] = bp[0];
            bb[t].f4[1] = bp[2];
        }
#pragma unroll
        for (int t = 0; t < 8; ++t)
            acc[t] = __builtin_amdgcn_wmma_f32_16x16x32_bf16(
                false, a.v, false, bb[t].v, (short)0, acc[t], false, false);
    }
#pragma unroll
    for (int t = 0; t < 8; ++t) {
        const int d = colBase + t * 16 + l16;
        const float bv = bias[d];
#pragma unroll
        for (int j = 0; j < 8; ++j) {
            const int r = j + half * 8;
            // cgate is re-read by 1024 workgroups right after: keep it cache-resident (RT)
            cout[(size_t)(rowBase + r) * D_MODEL + d] = acc[t][j] + bv;
        }
    }
}

// ---------------------------------------------------------------------------
// Kernel 3: fused  x = LN(leaky(input @ W_feat^T + b_feat) * c) * gamma + beta
// 1024 workgroups x 512 threads (16 waves). WG tile: 64 rows x 1024 cols.
// Wave tile: 2 row-tiles x 8 col-tiles (B frag reused x2).
// A slab (64x32 f32, 8KB) staged to LDS via global_load_async_to_lds_b128
// (non-temporal: input is a 128MB single-use stream), double buffered,
// cooperatively converted to a padded bf16 tile.
// ---------------------------------------------------------------------------
__global__ void __launch_bounds__(512)
feature_decoder_fused(const float* __restrict__ inp, const unsigned short* __restrict__ wbf,
                      const float* __restrict__ bfeat, const float* __restrict__ cgate,
                      const float* __restrict__ gamma, const float* __restrict__ beta,
                      float* __restrict__ out) {
    // f32 staging slab: 64 rows x 32 K, row stride 36 floats (pad -> conflict relief)
    __shared__ __align__(16) float Afs[2][64][36];
    // bf16 compute slab: 64 rows x 32 K, row stride 40 shorts (80B -> conflict-free frag reads)
    __shared__ __align__(16) unsigned short Abf[2][64][40];
    __shared__ float ps[64][8];
    __shared__ float pq[64][8];

    const int tid  = threadIdx.x;
    const int wave = tid >> 5, lane = tid & 31;
    const int half = lane >> 4, l16 = lane & 15;
    const int colGroup = wave & 7, rowGroup = wave >> 3;
    const int colBase  = colGroup * 128;
    const int rowBase  = blockIdx.x * 64;          // flat row (b*N + n)
    const int nBase    = rowBase & (N_SEQ - 1);    // row within sequence

    // per-thread async-copy chunk: one b128 (16B) of the 8KB slab
    const int cRow = tid >> 3;       // 0..63
    const int cSeg = tid & 7;        // 0..7  (16B segments of a 128B row)

    v8f acc[2][8];
#pragma unroll
    for (int m = 0; m < 2; ++m)
#pragma unroll
        for (int t = 0; t < 8; ++t) acc[m][t] = {};

    unsigned long long gsrc =
        (unsigned long long)(uintptr_t)(inp + (size_t)(rowBase + cRow) * N_FEAT + cSeg * 4);
    const unsigned int afsOff0 = (unsigned int)(uintptr_t)&Afs[0][cRow][cSeg * 4];
    const unsigned int afsOff1 = (unsigned int)(uintptr_t)&Afs[1][cRow][cSeg * 4];

    // ---- prologue: stage + convert slab 0
    asm volatile("global_load_async_to_lds_b128 %0, %1, off th:TH_LOAD_NT"
                 :: "v"(afsOff0), "v"(gsrc) : "memory");
    gsrc += 32u * sizeof(float);
    asm volatile("s_wait_asynccnt 0x0" ::: "memory");
    {
        float4 v = *(const float4*)&Afs[0][cRow][cSeg * 4];
        unsigned short* bdst = &Abf[0][cRow][cSeg * 4];
        bdst[0] = f2bf(v.x); bdst[1] = f2bf(v.y);
        bdst[2] = f2bf(v.z); bdst[3] = f2bf(v.w);
    }
    __syncthreads();

    for (int i = 0; i < 16; ++i) {
        const int nb = i + 1;
        // 1) kick off async prefetch of next slab (overlaps with compute below)
        if (nb < 16) {
            const unsigned int dst = (nb & 1) ? afsOff1 : afsOff0;
            asm volatile("global_load_async_to_lds_b128 %0, %1, off th:TH_LOAD_NT"
                         :: "v"(dst), "v"(gsrc) : "memory");
            gsrc += 32u * sizeof(float);
        }

        // 2) compute on current slab
        const int buf = i & 1;
        FragB af[2];
#pragma unroll
        for (int m = 0; m < 2; ++m) {
            const unsigned short* ap =
                &Abf[buf][rowGroup * 32 + m * 16 + l16][half * 8];
            af[m].f4[0] = *(const float4*)(ap);        // elems 0..7
            af[m].f4[1] = *(const float4*)(ap + 16);   // elems 8..15
        }
        const int kb = i * 32 + half * 8;
        FragB bb[8];
#pragma unroll
        for (int t = 0; t < 8; ++t) {
            const float4* bp =
                (const float4*)(wbf + (size_t)(colBase + t * 16 + l16) * N_FEAT + kb);
            bb[t].f4[0] = bp[0];
            bb[t].f4[1] = bp[2];
        }
#pragma unroll
        for (int t = 0; t < 8; ++t) {
            acc[0][t] = __builtin_amdgcn_wmma_f32_16x16x32_bf16(
                false, af[0].v, false, bb[t].v, (short)0, acc[0][t], false, false);
            acc[1][t] = __builtin_amdgcn_wmma_f32_16x16x32_bf16(
                false, af[1].v, false, bb[t].v, (short)0, acc[1][t], false, false);
        }

        // 3) land next slab + convert own chunk to bf16
        if (nb < 16) {
            asm volatile("s_wait_asynccnt 0x0" ::: "memory");
            float4 v = *(const float4*)&Afs[nb & 1][cRow][cSeg * 4];
            unsigned short* bdst = &Abf[nb & 1][cRow][cSeg * 4];
            bdst[0] = f2bf(v.x); bdst[1] = f2bf(v.y);
            bdst[2] = f2bf(v.z); bdst[3] = f2bf(v.w);
        }
        __syncthreads();
    }

    // ---- epilogue pass 1: bias + leaky-relu + gate; per-row partial moments
    float s[2][8], s2[2][8];
#pragma unroll
    for (int m = 0; m < 2; ++m)
#pragma unroll
        for (int j = 0; j < 8; ++j) { s[m][j] = 0.f; s2[m][j] = 0.f; }

#pragma unroll
    for (int m = 0; m < 2; ++m)
#pragma unroll
        for (int t = 0; t < 8; ++t) {
            const int d = colBase + t * 16 + l16;
            const float bv = bfeat[d];
#pragma unroll
            for (int j = 0; j < 8; ++j) {
                const int r = rowGroup * 32 + m * 16 + j + half * 8;
                float v = acc[m][t][j] + bv;
                v = (v >= 0.f) ? v : NEG_SLOPE * v;
                v *= cgate[(size_t)(nBase + r) * D_MODEL + d];
                acc[m][t][j] = v;
                s[m][j]  += v;
                s2[m][j] += v * v;
            }
        }
    // butterfly within each 16-lane half
#pragma unroll
    for (int m = 0; m < 2; ++m)
#pragma unroll
        for (int j = 0; j < 8; ++j) {
#pragma unroll
            for (int mk = 1; mk <= 8; mk <<= 1) {
                s[m][j]  += __shfl_xor(s[m][j],  mk, 32);
                s2[m][j] += __shfl_xor(s2[m][j], mk, 32);
            }
        }
    if (l16 == 0) {
#pragma unroll
        for (int m = 0; m < 2; ++m)
#pragma unroll
            for (int j = 0; j < 8; ++j) {
                const int r = rowGroup * 32 + m * 16 + j + half * 8;
                ps[r][colGroup] = s[m][j];
                pq[r][colGroup] = s2[m][j];
            }
    }
    __syncthreads();

    // ---- cross-wave reduce: full 1024-col row stats
    float mean[2][8], rstd[2][8];
#pragma unroll
    for (int m = 0; m < 2; ++m)
#pragma unroll
        for (int j = 0; j < 8; ++j) {
            const int r = rowGroup * 32 + m * 16 + j + half * 8;
            float su = 0.f, sq = 0.f;
#pragma unroll
            for (int w = 0; w < 8; ++w) { su += ps[r][w]; sq += pq[r][w]; }
            const float mu  = su * (1.0f / D_MODEL);
            const float var = sq * (1.0f / D_MODEL) - mu * mu;
            mean[m][j] = mu;
            rstd[m][j] = rsqrtf(var + LN_EPS);
        }

    // ---- epilogue pass 2: normalize + affine + non-temporal store (256MB stream)
#pragma unroll
    for (int m = 0; m < 2; ++m)
#pragma unroll
        for (int t = 0; t < 8; ++t) {
            const int d = colBase + t * 16 + l16;
            const float g  = gamma[d];
            const float be = beta[d];
#pragma unroll
            for (int j = 0; j < 8; ++j) {
                const int r = rowGroup * 32 + m * 16 + j + half * 8;
                float val = (acc[m][t][j] - mean[m][j]) * rstd[m][j] * g + be;
                __builtin_nontemporal_store(val, &out[(size_t)(rowBase + r) * D_MODEL + d]);
            }
        }
}

// ---------------------------------------------------------------------------
// Launch: inputs are (input, static, W_feat, b_feat, W_city, b_city, gamma, beta)
// Workspace: [0,1MB) W_feat bf16 | [1MB,1.5MB) W_city bf16 | [1.5MB,9.5MB) gate f32
// ---------------------------------------------------------------------------
extern "C" void kernel_launch(void* const* d_in, const int* in_sizes, int n_in,
                              void* d_out, int out_size, void* d_ws, size_t ws_size,
                              hipStream_t stream) {
    const float* inp   = (const float*)d_in[0];
    const float* statc = (const float*)d_in[1];
    const float* wfeat = (const float*)d_in[2];
    const float* bfeat = (const float*)d_in[3];
    const float* wcity = (const float*)d_in[4];
    const float* bcity = (const float*)d_in[5];
    const float* gamma = (const float*)d_in[6];
    const float* beta  = (const float*)d_in[7];
    float* out = (float*)d_out;

    char* ws = (char*)d_ws;
    unsigned short* wfeat_bf = (unsigned short*)ws;                          // 1 MB
    unsigned short* wcity_bf = (unsigned short*)(ws + (1u << 20));           // 0.5 MB
    float*          cgate    = (float*)(ws + (1u << 20) + (1u << 19));       // 8 MB

    const int nWF = D_MODEL * N_FEAT;   // 524288
    const int nWC = D_MODEL * N_CLUST;  // 262144
    convert_f32_bf16<<<(nWF + 255) / 256, 256, 0, stream>>>(wfeat, wfeat_bf, nWF);
    convert_f32_bf16<<<(nWC + 255) / 256, 256, 0, stream>>>(wcity, wcity_bf, nWC);

    city_gemm<<<N_SEQ / 16, 256, 0, stream>>>(statc, wcity_bf, bcity, cgate);

    feature_decoder_fused<<<(N_BATCH * N_SEQ) / 64, 512, 0, stream>>>(
        inp, wfeat_bf, bfeat, cgate, gamma, beta, out);
}
